// CausalRpeAttn_75076028334342
// MI455X (gfx1250) — matrix operand order repaired and checked by reading, not compiled, not run on hardware
//
#include <hip/hip_runtime.h>
#include <math.h>

#define BATCH  2
#define SEQ    2048
#define DMODEL 1024
#define NHEADS 16
#define NKV    4
#define DHEAD  64
#define GROUP  4
#define NROWS  (BATCH*SEQ)   /* 4096 */
#define NQKV   1536          /* 1024 + 256 + 256 */
#define SCALE_F 0.125f       /* 1/sqrt(64) */

typedef __attribute__((ext_vector_type(16))) __bf16 v16bf;
typedef __attribute__((ext_vector_type(2)))  __bf16 v2bf;
typedef __attribute__((ext_vector_type(8)))  float  v8f;

__device__ __forceinline__ unsigned short f2bf(float f) {
  __bf16 h = (__bf16)f;
  return __builtin_bit_cast(unsigned short, h);
}
__device__ __forceinline__ v2bf u2bf2(unsigned int u) {
  return __builtin_bit_cast(v2bf, u);
}
__device__ __forceinline__ v8f zero8() {
  v8f z;
  #pragma unroll
  for (int i = 0; i < 8; i++) z[i] = 0.0f;
  return z;
}
__device__ __forceinline__ v8f wmma_bf16(v16bf a, v16bf b, v8f c) {
  // D = A(16x32 bf16) * B(32x16 bf16) + C(16x16 f32)
  return __builtin_amdgcn_wmma_f32_16x16x32_bf16(
      /*neg_a=*/false, a, /*neg_b=*/false, b,
      /*c_mod=*/(short)0, c, /*reuse_a=*/false, /*reuse_b=*/false);
}

// ---- CDNA5 async global->LDS DMA (ASYNCcnt-tracked, ISA ch.10 async ops) ----
__device__ __forceinline__ unsigned lds_off(const void* p) {
  // Flat LDS address: LDS offset is the low 32 bits (ISA 10.2 aperture mapping)
  return (unsigned)(unsigned long long)p;
}
__device__ __forceinline__ void async_b128(unsigned lds_byte, unsigned gvoff_byte,
                                           const void* sbase) {
  asm volatile("global_load_async_to_lds_b128 %0, %1, %2"
               :: "v"(lds_byte), "v"(gvoff_byte), "s"(sbase) : "memory");
}
__device__ __forceinline__ void wait_async0() {
  asm volatile("s_wait_asynccnt 0" ::: "memory");
}

// ---------------------------------------------------------------------------
// Elementwise convert / pack kernels
// ---------------------------------------------------------------------------
__global__ void __launch_bounds__(256)
f32_to_bf16_k(const float* __restrict__ src, unsigned short* __restrict__ dst, int n) {
  int i = blockIdx.x * 256 + threadIdx.x;
  if (i < n) dst[i] = f2bf(src[i]);
}

__global__ void __launch_bounds__(256)
pack_w_k(const float* __restrict__ src, unsigned short* __restrict__ dst,
         int rows, int cols, int dstld, int dstoff) {
  int i = blockIdx.x * 256 + threadIdx.x;
  if (i < rows * cols) {
    int r = i / cols, c = i % cols;
    dst[(size_t)r * dstld + dstoff + c] = f2bf(src[i]);
  }
}

__global__ void __launch_bounds__(256)
pack_bias_k(const float* __restrict__ bq, const float* __restrict__ bk,
            const float* __restrict__ bv, float* __restrict__ bcat) {
  int i = blockIdx.x * 256 + threadIdx.x;
  if (i < 1024)      bcat[i] = bq[i];
  else if (i < 1280) bcat[i] = bk[i - 1024];
  else if (i < 1536) bcat[i] = bv[i - 1280];
}

// ---------------------------------------------------------------------------
// WMMA GEMM: C[M,N] = A[M,K](bf16) * B[K,N](bf16) + bias[N], C fp32.
// 128x128 block tile, BK=32, 256 threads = 8 waves, each wave: 64x32 output.
// A tile staged by async DMA (no VGPR round-trip); B tile transposed via VGPRs.
// ---------------------------------------------------------------------------
__global__ void __launch_bounds__(256)
gemm_bf16_bias(const unsigned short* __restrict__ A,
               const unsigned short* __restrict__ Bw,
               const float* __restrict__ bias,
               float* __restrict__ C, int M, int N, int K) {
  __shared__ __align__(16) unsigned short sA[128 * 32];   // [row][k]
  __shared__ __align__(16) unsigned short sB[128 * 32];   // [col][k] (transposed)
  const int tid  = threadIdx.x;
  const int bm   = blockIdx.y * 128, bn = blockIdx.x * 128;
  const int wave = tid >> 5, lane = tid & 31;
  const int wm   = (wave & 1) * 64;       // wave row offset in tile
  const int wn   = (wave >> 1) * 32;      // wave col offset in tile
  const int mA   = lane & 15, koA = (lane >> 4) * 8;   // A-frag layout (§7.12.2)
  const int nB   = lane & 15, koB = (lane >> 4) * 8;   // B-frag layout

  const int arow = tid >> 1, aoff = (tid & 1) * 16;    // A-tile loader mapping
  const int bkr  = tid >> 3, bcs = (tid & 7) * 16;     // B-tile loader mapping

  v8f acc[4][2];
  #pragma unroll
  for (int i = 0; i < 4; i++)
    #pragma unroll
    for (int j = 0; j < 2; j++) acc[i][j] = zero8();

  for (int k0 = 0; k0 < K; k0 += 32) {
    __syncthreads();
    // Prefetch next iteration's tiles into L2/WGP$ while this one computes.
    if (k0 + 32 < K) {
      __builtin_prefetch(A  + (size_t)(bm + arow) * K + k0 + 32 + aoff, 0, 1);
      __builtin_prefetch(Bw + (size_t)(k0 + 32 + bkr) * N + bn + bcs, 0, 1);
    }
    { // A tile 128x32: 2x b128 async DMA straight into LDS per thread
      unsigned gb = (unsigned)(((unsigned)(bm + arow) * (unsigned)K +
                                (unsigned)(k0 + aoff)) * 2u);
      unsigned lb = lds_off(sA + arow * 32 + aoff);
      async_b128(lb,      gb,      A);
      async_b128(lb + 16, gb + 16, A);
    }
    { // B tile 32x128, coalesced global read, transposed LDS write
      const uint4* src = (const uint4*)(Bw + (size_t)(k0 + bkr) * N + bn + bcs);
      uint4 d0 = src[0], d1 = src[1];
      unsigned short tmp[16];
      *(uint4*)(tmp) = d0; *(uint4*)(tmp + 8) = d1;
      #pragma unroll
      for (int c = 0; c < 16; c++) sB[(bcs + c) * 32 + bkr] = tmp[c];
    }
    wait_async0();
    __syncthreads();

    v16bf af[4];
    #pragma unroll
    for (int i = 0; i < 4; i++) {
      const unsigned short* base = sA + (wm + i * 16 + mA) * 32;
      #pragma unroll
      for (int v = 0; v < 8; v++) {
        int kb = ((v < 4) ? 0 : 16) + koA + 2 * (v & 3);
        v2bf p = u2bf2(*(const unsigned int*)(base + kb));
        af[i][2 * v] = p[0]; af[i][2 * v + 1] = p[1];
      }
    }
    v16bf bf[2];
    #pragma unroll
    for (int j = 0; j < 2; j++) {
      const unsigned short* base = sB + (wn + j * 16 + nB) * 32 + koB;
      #pragma unroll
      for (int v = 0; v < 8; v++) {
        v2bf p = u2bf2(*(const unsigned int*)(base + ((v < 4) ? 0 : 16) + 2 * (v & 3)));
        bf[j][2 * v] = p[0]; bf[j][2 * v + 1] = p[1];
      }
    }
    #pragma unroll
    for (int i = 0; i < 4; i++)
      #pragma unroll
      for (int j = 0; j < 2; j++)
        acc[i][j] = wmma_bf16(af[i], bf[j], acc[i][j]);
  }

  // Epilogue: C layout row m = (lane>>4)*8+v, col n = lane&15
  #pragma unroll
  for (int i = 0; i < 4; i++)
    #pragma unroll
    for (int j = 0; j < 2; j++)
      #pragma unroll
      for (int v = 0; v < 8; v++) {
        int row = bm + wm + i * 16 + (lane >> 4) * 8 + v;
        int col = bn + wn + j * 16 + (lane & 15);
        C[(size_t)row * N + col] = acc[i][j][v] + bias[col];
      }
}

// ---------------------------------------------------------------------------
// RoPE + split into per-head bf16 buffers: Q[B,H,S,64], K[B,KV,S,64], V[B,KV,S,64]
// ---------------------------------------------------------------------------
__global__ void __launch_bounds__(256)
rope_split(const float* __restrict__ QKV, const float* __restrict__ pos,
           unsigned short* __restrict__ Qb, unsigned short* __restrict__ Kb,
           unsigned short* __restrict__ Vb) {
  int r = blockIdx.x;                 // 0..NROWS-1
  int b = r / SEQ, s = r % SEQ;
  const float* rowp = QKV + (size_t)r * NQKV;
  for (int c = threadIdx.x; c < NQKV; c += 256) {
    if (c < 1024) {
      int h = c >> 6, d = c & 63, dm = d & 31;
      float ang = pos[s * 32 + dm];
      float cs = cosf(ang), sn = sinf(ang);
      float x = rowp[c];
      float y = rowp[(h << 6) + ((d < 32) ? d + 32 : d - 32)];
      float o = (d < 32) ? (x * cs - y * sn) : (x * cs + y * sn);
      Qb[(((size_t)b * NHEADS + h) * SEQ + s) * DHEAD + d] = f2bf(o);
    } else if (c < 1280) {
      int c2 = c - 1024, kv = c2 >> 6, d = c2 & 63, dm = d & 31;
      float ang = pos[s * 32 + dm];
      float cs = cosf(ang), sn = sinf(ang);
      float x = rowp[c];
      float y = rowp[1024 + (kv << 6) + ((d < 32) ? d + 32 : d - 32)];
      float o = (d < 32) ? (x * cs - y * sn) : (x * cs + y * sn);
      Kb[(((size_t)b * NKV + kv) * SEQ + s) * DHEAD + d] = f2bf(o);
    } else {
      int c2 = c - 1280, kv = c2 >> 6, d = c2 & 63;
      Vb[(((size_t)b * NKV + kv) * SEQ + s) * DHEAD + d] = f2bf(rowp[c]);
    }
  }
}

// ---------------------------------------------------------------------------
// Causal flash attention, WMMA bf16. Block = 128 threads = 4 waves, 64 q-rows.
// Each wave owns 16 q-rows; K tile DMA'd async into LDS, V transposed via VGPRs.
// Output Ob: bf16 [B*S, H*64] (row-major, ready for the O-projection GEMM).
// ---------------------------------------------------------------------------
__global__ void __launch_bounds__(128)
flash_attn(const unsigned short* __restrict__ Qb,
           const unsigned short* __restrict__ Kb,
           const unsigned short* __restrict__ Vb,
           unsigned short* __restrict__ Ob) {
  __shared__ __align__(16) unsigned short sK[32 * 64];     // [key][d]
  __shared__ __align__(16) unsigned short sV[64 * 32];     // [d][key] transposed
  __shared__ __align__(16) unsigned short sP[4][16 * 32];  // per-wave P tile

  const int QTILES = SEQ / 64;
  int idx = blockIdx.x;
  int qt = idx % QTILES; idx /= QTILES;
  int h  = idx % NHEADS; int b = idx / NHEADS;
  int hkv = h / GROUP;
  const unsigned short* Qh = Qb + ((size_t)b * NHEADS + h)   * SEQ * DHEAD;
  const unsigned short* Kh = Kb + ((size_t)b * NKV    + hkv) * SEQ * DHEAD;
  const unsigned short* Vh = Vb + ((size_t)b * NKV    + hkv) * SEQ * DHEAD;

  int wave = threadIdx.x >> 5, lane = threadIdx.x & 31;
  int qbase = qt * 64 + wave * 16;
  int mA = lane & 15, koA = (lane >> 4) * 8;
  int nB = lane & 15, koB = (lane >> 4) * 8;
  const int lk = threadIdx.x >> 2, lds_ = (threadIdx.x & 3) * 16;  // K/V loader map

  // Q fragments (16 rows x 64 d = two 16x32 A-frags), read directly from global
  v16bf qf[2];
  #pragma unroll
  for (int t = 0; t < 2; t++)
    #pragma unroll
    for (int v = 0; v < 8; v++) {
      int kb = t * 32 + ((v < 4) ? 0 : 16) + koA + 2 * (v & 3);
      v2bf p = u2bf2(*(const unsigned int*)(Qh + (size_t)(qbase + mA) * DHEAD + kb));
      qf[t][2 * v] = p[0]; qf[t][2 * v + 1] = p[1];
    }

  float Mx[8], Ls[8];
  #pragma unroll
  for (int v = 0; v < 8; v++) { Mx[v] = -__builtin_inff(); Ls[v] = 0.0f; }
  v8f acc[4];
  #pragma unroll
  for (int t = 0; t < 4; t++) acc[t] = zero8();

  int jend = qt * 64 + 63;
  for (int j = 0; j <= jend; j += 32) {
    __syncthreads();
    if (j + 32 <= jend) {  // prefetch next K/V tile
      __builtin_prefetch(Kh + (size_t)(j + 32 + lk) * DHEAD + lds_, 0, 1);
      __builtin_prefetch(Vh + (size_t)(j + 32 + lk) * DHEAD + lds_, 0, 1);
    }
    { // K tile 32x64: async DMA straight into LDS (2x b128 per thread)
      unsigned gb = (unsigned)(((unsigned)(j + lk) * DHEAD + lds_) * 2u);
      unsigned lb = lds_off(sK + lk * 64 + lds_);
      async_b128(lb,      gb,      Kh);
      async_b128(lb + 16, gb + 16, Kh);
    }
    { // V tile: coalesced read + transposed LDS write ([d][key])
      const uint4* vsrc = (const uint4*)(Vh + (size_t)(j + lk) * DHEAD + lds_);
      uint4 v0 = vsrc[0], v1 = vsrc[1];
      unsigned short tmp[16];
      *(uint4*)(tmp) = v0; *(uint4*)(tmp + 8) = v1;
      #pragma unroll
      for (int c = 0; c < 16; c++) sV[(lds_ + c) * 32 + lk] = tmp[c];
    }
    wait_async0();
    __syncthreads();

    if (j <= qbase + 15) {   // this wave still has unmasked keys in this tile
      // ---- scores: S[16q x 32k] as two 16x16 tiles, K-dim = Dh = 2 chunks ----
      v8f sc[2];
      #pragma unroll
      for (int tt = 0; tt < 2; tt++) {
        sc[tt] = zero8();
        #pragma unroll
        for (int t = 0; t < 2; t++) {
          v16bf kf;   // B = K^T chunk: B[d][key] = K[key][d], pairs contiguous in sK
          #pragma unroll
          for (int v = 0; v < 8; v++) {
            int d = t * 32 + ((v < 4) ? 0 : 16) + koB + 2 * (v & 3);
            v2bf p = u2bf2(*(const unsigned int*)(sK + (tt * 16 + nB) * 64 + d));
            kf[2 * v] = p[0]; kf[2 * v + 1] = p[1];
          }
          sc[tt] = wmma_bf16(qf[t], kf, sc[tt]);
        }
      }
      // ---- online softmax (row = (lane>>4)*8+v, col = lane&15) ----
      #pragma unroll
      for (int v = 0; v < 8; v++) {
        int qrow = qbase + (lane >> 4) * 8 + v;
        float e0 = sc[0][v] * SCALE_F, e1 = sc[1][v] * SCALE_F;
        int k0i = j + (lane & 15), k1i = k0i + 16;
        if (k0i > qrow) e0 = -__builtin_inff();
        if (k1i > qrow) e1 = -__builtin_inff();
        float r = fmaxf(e0, e1);
        #pragma unroll
        for (int o = 8; o >= 1; o >>= 1) r = fmaxf(r, __shfl_xor(r, o, 32));
        float newM = fmaxf(fmaxf(Mx[v], r), -1e30f);   // clamp: all-masked -> zeros
        float corr = __expf(Mx[v] - newM);
        float p0 = __expf(e0 - newM), p1 = __expf(e1 - newM);
        float rs = p0 + p1;
        #pragma unroll
        for (int o = 8; o >= 1; o >>= 1) rs += __shfl_xor(rs, o, 32);
        Ls[v] = Ls[v] * corr + rs;
        Mx[v] = newM;
        #pragma unroll
        for (int t = 0; t < 4; t++) acc[t][v] *= corr;
        int m = (lane >> 4) * 8 + v;
        sP[wave][m * 32 + (lane & 15)]      = f2bf(p0);
        sP[wave][m * 32 + 16 + (lane & 15)] = f2bf(p1);
      }
      // ---- P (16x32) @ V (32x64): re-fragment P from LDS into A layout ----
      v16bf pf;
      #pragma unroll
      for (int v = 0; v < 8; v++) {
        int kb = ((v < 4) ? 0 : 16) + koA + 2 * (v & 3);
        v2bf p = u2bf2(*(const unsigned int*)(&sP[wave][mA * 32 + kb]));
        pf[2 * v] = p[0]; pf[2 * v + 1] = p[1];
      }
      #pragma unroll
      for (int t = 0; t < 4; t++) {
        v16bf vf;   // B = V chunk: B[key][d], key pairs contiguous in transposed sV
        #pragma unroll
        for (int v = 0; v < 8; v++) {
          v2bf p = u2bf2(*(const unsigned int*)(sV + (t * 16 + nB) * 32 + ((v < 4) ? 0 : 16) + koB + 2 * (v & 3)));
          vf[2 * v] = p[0]; vf[2 * v + 1] = p[1];
        }
        acc[t] = wmma_bf16(pf, vf, acc[t]);
      }
    }
  }

  // Epilogue: normalize and store bf16 at [b*S + q, h*64 + d]
  #pragma unroll
  for (int v = 0; v < 8; v++) {
    float inv = 1.0f / Ls[v];     // diagonal always unmasked -> Ls > 0
    int srow = qbase + (lane >> 4) * 8 + v;
    #pragma unroll
    for (int t = 0; t < 4; t++) {
      int col = h * DHEAD + t * 16 + (lane & 15);
      Ob[(size_t)(b * SEQ + srow) * DMODEL + col] = f2bf(acc[t][v] * inv);
    }
  }
}

// ---------------------------------------------------------------------------
// Host-side launcher
// ---------------------------------------------------------------------------
extern "C" void kernel_launch(void* const* d_in, const int* in_sizes, int n_in,
                              void* d_out, int out_size, void* d_ws, size_t ws_size,
                              hipStream_t stream) {
  (void)in_sizes; (void)n_in; (void)out_size; (void)ws_size;
  const float* qkv = (const float*)d_in[0];
  const float* pos = (const float*)d_in[1];
  const float* Wq  = (const float*)d_in[2];
  const float* bq  = (const float*)d_in[3];
  const float* Wk  = (const float*)d_in[4];
  const float* bk  = (const float*)d_in[5];
  const float* Wv  = (const float*)d_in[6];
  const float* bv  = (const float*)d_in[7];
  const float* Wo  = (const float*)d_in[8];
  const float* bo  = (const float*)d_in[9];
  float* out = (float*)d_out;
  char* ws = (char*)d_ws;

  size_t off = 0;
  auto take = [&](size_t bytes) { char* p = ws + off; off += (bytes + 255) & ~(size_t)255; return p; };
  unsigned short* Xb   = (unsigned short*)take((size_t)NROWS * DMODEL * 2);  // 8 MB
  unsigned short* Wcat = (unsigned short*)take((size_t)DMODEL * NQKV * 2);   // 3 MB
  unsigned short* Wob  = (unsigned short*)take((size_t)DMODEL * DMODEL * 2); // 2 MB
  float*          bcat = (float*)         take((size_t)NQKV * 4);
  float*          QKVc = (float*)         take((size_t)NROWS * NQKV * 4);    // 24 MB
  unsigned short* Qbuf = (unsigned short*)take((size_t)BATCH * NHEADS * SEQ * DHEAD * 2); // 8 MB
  unsigned short* Kbuf = (unsigned short*)take((size_t)BATCH * NKV * SEQ * DHEAD * 2);    // 2 MB
  unsigned short* Vbuf = (unsigned short*)take((size_t)BATCH * NKV * SEQ * DHEAD * 2);    // 2 MB
  unsigned short* Attn = (unsigned short*)take((size_t)NROWS * DMODEL * 2);  // 8 MB

  // 1. Convert activations + weights to bf16 (pack Wq|Wk|Wv into one matrix)
  f32_to_bf16_k<<<(NROWS * DMODEL + 255) / 256, 256, 0, stream>>>(qkv, Xb, NROWS * DMODEL);
  pack_w_k<<<(1024 * 1024 + 255) / 256, 256, 0, stream>>>(Wq, Wcat, 1024, 1024, NQKV, 0);
  pack_w_k<<<(1024 * 256  + 255) / 256, 256, 0, stream>>>(Wk, Wcat, 1024, 256,  NQKV, 1024);
  pack_w_k<<<(1024 * 256  + 255) / 256, 256, 0, stream>>>(Wv, Wcat, 1024, 256,  NQKV, 1280);
  pack_w_k<<<(1024 * 1024 + 255) / 256, 256, 0, stream>>>(Wo, Wob, 1024, 1024, 1024, 0);
  pack_bias_k<<<6, 256, 0, stream>>>(bq, bk, bv, bcat);

  // 2. Fused QKV projection (WMMA): [4096,1024] @ [1024,1536] + bias
  gemm_bf16_bias<<<dim3(NQKV / 128, NROWS / 128), 256, 0, stream>>>(
      Xb, Wcat, bcat, QKVc, NROWS, NQKV, DMODEL);

  // 3. RoPE + split to per-head bf16 layouts
  rope_split<<<NROWS, 256, 0, stream>>>(QKVc, pos, Qbuf, Kbuf, Vbuf);

  // 4. Causal GQA flash attention (WMMA + async-LDS K staging)
  flash_attn<<<BATCH * NHEADS * (SEQ / 64), 128, 0, stream>>>(Qbuf, Kbuf, Vbuf, Attn);

  // 5. Output projection (WMMA): [4096,1024] @ [1024,1024] + bo -> fp32 out
  gemm_bf16_bias<<<dim3(DMODEL / 128, NROWS / 128), 256, 0, stream>>>(
      Attn, Wob, bo, out, NROWS, DMODEL, DMODEL);
}
